// MMLinearP_25254407700651
// MI455X (gfx1250) — matrix operand order repaired and checked by reading, not compiled, not run on hardware
//
#include <hip/hip_runtime.h>
#include <hip/hip_bf16.h>

typedef __attribute__((ext_vector_type(16))) _Float16 v16h;
typedef __attribute__((ext_vector_type(8)))  float    v8f;

#define N_TOK 8192
#define LEN   512
#define NE    8

// workspace layout (bytes)
#define WS_COUNTS 0
#define WS_RG     256
#define WS_GM     512
#define WS_BET    1024                       // 8*512*4 = 16384
#define WS_IDX    20480                      // 8*8192*4 = 262144
#define WS_WGT    282624                     // 8*8192*4 = 262144
#define WS_BPACK  544768                     // 8*512*512*2 = 4194304 (32B aligned)

// ---------------- out = 0 ----------------
__global__ void zero_out_k(float* __restrict__ out) {
  int i = blockIdx.x * blockDim.x + threadIdx.x;         // over 1,048,576 float4
  ((float4*)out)[i] = make_float4(0.f, 0.f, 0.f, 0.f);
}

// ---------------- tiny stats: ibar -> router_gamma, gam, bet ----------------
__global__ void init_params_k(const float* __restrict__ ins,
                              const float* __restrict__ rmod_w,
                              const float* __restrict__ gamma_w,
                              const float* __restrict__ beta_w,
                              float* __restrict__ rg, float* __restrict__ gm,
                              float* __restrict__ bet, int* __restrict__ counts) {
  __shared__ float ibar[LEN];
  int t = threadIdx.x;                                   // 512 threads
  if (t < NE) counts[t] = 0;
  float s = 0.f;
  #pragma unroll
  for (int n = 0; n < 16; ++n) s += ins[n * LEN + t];
  ibar[t] = s * (1.f / 16.f);
  __syncthreads();
  if (t < NE) {
    float a = 0.f, b = 0.f;
    for (int h = 0; h < LEN; ++h) {
      float iv = ibar[h];
      a += iv * rmod_w[h * NE + t];                      // rmod_w [512,8]
      b += iv * gamma_w[t * LEN + h];                    // gamma_w [8,512]
    }
    rg[t] = a;
    gm[t] = b;
  }
  for (int e = 0; e < NE; ++e) {                         // bet[e,d] = ibar . beta_w[e,:,d]
    float s2 = 0.f;
    for (int h = 0; h < LEN; ++h) s2 += ibar[h] * beta_w[(e * LEN + h) * LEN + t];
    bet[e * LEN + t] = s2;
  }
}

// ---------------- routing: one wave per token ----------------
__global__ void route_k(const float* __restrict__ x, const float* __restrict__ gate_w,
                        const float* __restrict__ rg, int* __restrict__ counts,
                        int* __restrict__ idx, float* __restrict__ wgt) {
  int lane  = threadIdx.x & 31;
  int token = blockIdx.x * 8 + (threadIdx.x >> 5);       // 1024 blocks * 8 waves
  const float* xr = x + token * LEN;
  float p[NE];
  #pragma unroll
  for (int e = 0; e < NE; ++e) p[e] = 0.f;
  for (int h = lane; h < LEN; h += 32) {
    float xv = xr[h];
    const float* g = gate_w + h * NE;                    // gate_w [512,8] row-contig
    #pragma unroll
    for (int e = 0; e < NE; ++e) p[e] += xv * g[e];
  }
  #pragma unroll
  for (int off = 16; off; off >>= 1) {
    #pragma unroll
    for (int e = 0; e < NE; ++e) p[e] += __shfl_xor(p[e], off, 32);
  }
  if (lane == 0) {
    float l[NE];
    float m = -1e30f;
    #pragma unroll
    for (int e = 0; e < NE; ++e) { l[e] = p[e] + rg[e]; m = fmaxf(m, l[e]); }
    float s = 0.f;
    #pragma unroll
    for (int e = 0; e < NE; ++e) { l[e] = expf(l[e] - m); s += l[e]; }
    float inv = 1.f / s;
    int i0 = 0; float v0 = l[0];
    #pragma unroll
    for (int e = 1; e < NE; ++e) if (l[e] > v0) { v0 = l[e]; i0 = e; }   // first-index ties
    int i1 = -1; float v1 = -1.f;
    #pragma unroll
    for (int e = 0; e < NE; ++e) if (e != i0 && l[e] > v1) { v1 = l[e]; i1 = e; }
    int s0 = atomicAdd(&counts[i0], 1);
    idx[i0 * N_TOK + s0] = token; wgt[i0 * N_TOK + s0] = v0 * inv;
    int s1 = atomicAdd(&counts[i1], 1);
    idx[i1 * N_TOK + s1] = token; wgt[i1 * N_TOK + s1] = v1 * inv;
  }
}

// ---------------- repack expert_w (f32) -> WMMA B fragments (f16) ----------------
// Fragment f = (e*32 + colChunk)*16 + kStep ; lane holds 16 halfs: col = cc*16 + (lane&15),
// K = kStep*32 + (lane>>4)*16 + i   (contiguous K order, same as the A fragment)
__global__ void repack_k(const float* __restrict__ ew, _Float16* __restrict__ bp) {
  int g    = blockIdx.x * 8 + (threadIdx.x >> 5);        // 4096 fragments
  int lane = threadIdx.x & 31;
  int e    = g >> 9;
  int rem  = g & 511;
  int cc   = rem >> 4;                                   // col chunk 0..31
  int kt   = rem & 15;                                   // K step 0..15
  int col  = cc * 16 + (lane & 15);
  int kb   = kt * 32 + (lane >> 4) * 16;
  const float* src = ew + (e * LEN + kb) * LEN + col;    // expert_w [E,512,512]
  v16h v;
  #pragma unroll
  for (int i = 0; i < 16; ++i) v[i] = (_Float16)src[i * LEN];
  *(v16h*)(bp + ((size_t)g * 32 + lane) * 16) = v;
}

// ---------------- grouped MoE GEMM ----------------
// Block: 256 thr = 8 waves = 4 row-groups x 2 col-halves; tile = 64 rows x 256 cols.
// Row-group waves share identical B addresses (WGP$ dedup); col-half waves share A.
// grid = (128 row tiles, 8 experts, 2 col halves)
__global__ __launch_bounds__(256) void moe_gemm_k(
    const float* __restrict__ x, const _Float16* __restrict__ bp,
    const int* __restrict__ counts, const int* __restrict__ idx,
    const float* __restrict__ wgt, const float* __restrict__ gm,
    const float* __restrict__ bet, const float* __restrict__ eb,
    float* __restrict__ out) {
  int e    = blockIdx.y;
  int tile = blockIdx.x;
  int cnt  = counts[e];
  if (tile * 64 >= cnt) return;

  __shared__ int   tokS[64];
  __shared__ float wS[64];
  int t = threadIdx.x;
  if (t < 64) {
    int slot = tile * 64 + t;
    if (slot < cnt) { tokS[t] = idx[e * N_TOK + slot]; wS[t] = wgt[e * N_TOK + slot]; }
    else            { tokS[t] = 0;                      wS[t] = 0.f; }
  }
  __syncthreads();

  int wave = t >> 5, lane = t & 31;
  int rg      = wave >> 1;                               // row group 0..3
  int rbase   = rg * 16;
  int colbase = blockIdx.z * 256 + (wave & 1) * 128;     // this wave: 128 cols
  int r   = rbase + (lane & 15);                         // A row for this lane
  int kbA = (lane >> 4) * 16;                            // A-fragment K sub-base
  const float* xrow = x + tokS[r] * LEN;

  v8f acc[8] = {};
  const _Float16* bbase = bp + (size_t)((e * 32 + (colbase >> 4)) * 16) * 512;

  for (int kt = 0; kt < 16; ++kt) {
    // A fragment: elems 0..15 -> K = kt*32 + kbA + i
    v16h a;
    const float* pA = xrow + kt * 32 + kbA;
    #pragma unroll
    for (int i = 0; i < 8; ++i) a[i] = (_Float16)pA[i];
    #pragma unroll
    for (int i = 0; i < 8; ++i) a[8 + i] = (_Float16)pA[8 + i];
    #pragma unroll
    for (int c = 0; c < 8; ++c) {
      const v16h b = *(const v16h*)(bbase + (size_t)(c * 16 + kt) * 512 + lane * 16);
      acc[c] = __builtin_amdgcn_wmma_f32_16x16x32_f16(
          false, a, false, b, (short)0, acc[c], false, false);
    }
  }

  // Epilogue: val = w * (gam[e]*(acc + expert_b[e,col]) + bet[e,col]); atomic add into out.
  float gme = gm[e];
  int lh = lane >> 4, lc = lane & 15;
  float ebv[8], btv[8];
  #pragma unroll
  for (int c = 0; c < 8; ++c) {
    int col = colbase + c * 16 + lc;
    ebv[c] = eb[e * LEN + col];
    btv[c] = bet[e * LEN + col];
  }
  #pragma unroll
  for (int j = 0; j < 8; ++j) {
    int m    = rbase + j + 8 * lh;                       // C/D layout: M = j + 8*(lane>>4)
    int slot = tile * 64 + m;
    if (slot >= cnt) continue;
    float w = wS[m];
    float* orow = out + tokS[m] * LEN + colbase + lc;
    #pragma unroll
    for (int c = 0; c < 8; ++c) {
      float val = w * (gme * (acc[c][j] + ebv[c]) + btv[c]);
      atomicAdd(orow + c * 16, val);
    }
  }
}

extern "C" void kernel_launch(void* const* d_in, const int* in_sizes, int n_in,
                              void* d_out, int out_size, void* d_ws, size_t ws_size,
                              hipStream_t stream) {
  (void)in_sizes; (void)n_in; (void)out_size; (void)ws_size;
  const float* x        = (const float*)d_in[0];
  const float* ins      = (const float*)d_in[1];
  const float* gate_w   = (const float*)d_in[2];
  const float* expert_w = (const float*)d_in[3];
  const float* expert_b = (const float*)d_in[4];
  const float* gamma_w  = (const float*)d_in[5];
  const float* beta_w   = (const float*)d_in[6];
  const float* rmod_w   = (const float*)d_in[7];
  float* out = (float*)d_out;

  char* ws = (char*)d_ws;
  int*      counts = (int*)(ws + WS_COUNTS);
  float*    rg     = (float*)(ws + WS_RG);
  float*    gm     = (float*)(ws + WS_GM);
  float*    bet    = (float*)(ws + WS_BET);
  int*      idx    = (int*)(ws + WS_IDX);
  float*    wgt    = (float*)(ws + WS_WGT);
  _Float16* bp     = (_Float16*)(ws + WS_BPACK);

  zero_out_k<<<4096, 256, 0, stream>>>(out);                       // 4,194,304 floats
  init_params_k<<<1, 512, 0, stream>>>(ins, rmod_w, gamma_w, beta_w, rg, gm, bet, counts);
  route_k<<<1024, 256, 0, stream>>>(x, gate_w, rg, counts, idx, wgt);
  repack_k<<<512, 256, 0, stream>>>(expert_w, bp);
  dim3 g(N_TOK / 64, NE, 2);
  moe_gemm_k<<<g, 256, 0, stream>>>(x, bp, counts, idx, wgt, gm, bet, expert_b, out);
}
